// GraphEncoder_65816078844234
// MI455X (gfx1250) — matrix operand order repaired and checked by reading, not compiled, not run on hardware
//
#include <hip/hip_runtime.h>
#include <hip/hip_bf16.h>

#define NNODES 100000
#define NEDGES 1600000
#define NPAD   100032   // NNODES rounded up to a multiple of 64 (GEMM row tile)

typedef __attribute__((ext_vector_type(16))) _Float16 v16h;
typedef __attribute__((ext_vector_type(8)))  _Float16 v8h;
typedef __attribute__((ext_vector_type(8)))  float    v8f;

// ---------------------------------------------------------------- utilities

__global__ __launch_bounds__(256) void zero_f32_kernel(float* __restrict__ p, long long n) {
    long long stride = (long long)gridDim.x * blockDim.x;
    for (long long i = (long long)blockIdx.x * blockDim.x + threadIdx.x; i < n; i += stride)
        p[i] = 0.0f;
}

__global__ __launch_bounds__(256) void count_kernel(const int* __restrict__ dst,
                                                    float* __restrict__ cnt, int e) {
    int stride = gridDim.x * blockDim.x;
    for (int i = blockIdx.x * blockDim.x + threadIdx.x; i < e; i += stride)
        atomicAdd(&cnt[dst[i]], 1.0f);
}

__global__ __launch_bounds__(256) void inv_count_kernel(float* __restrict__ cnt, int n) {
    int stride = gridDim.x * blockDim.x;
    for (int i = blockIdx.x * blockDim.x + threadIdx.x; i < n; i += stride)
        cnt[i] = 1.0f / fmaxf(cnt[i], 1.0f);
}

// scatter-add of neighbor messages: msg[dst] += x[src]  (global_atomic_add_f32)
__global__ __launch_bounds__(256) void scatter_add_kernel(const float* __restrict__ x,
                                                          const int* __restrict__ src,
                                                          const int* __restrict__ dst,
                                                          float* __restrict__ msg,
                                                          long long total, int d) {
    long long stride = (long long)gridDim.x * blockDim.x;
    for (long long i = (long long)blockIdx.x * blockDim.x + threadIdx.x; i < total; i += stride) {
        int e = (int)(i / d);
        int k = (int)(i - (long long)e * d);
        int s = src[e];
        int t = dst[e];
        atomicAdd(&msg[(size_t)t * d + k], x[(size_t)s * d + k]);
    }
}

// Build f16 staging matrix A = [mean | x], row-major [NPAD x Kpad], zero padded.
__global__ __launch_bounds__(256) void build_a_kernel(const float* __restrict__ msg,
                                                      const float* __restrict__ x,
                                                      const float* __restrict__ inv,
                                                      _Float16* __restrict__ Ah,
                                                      int n, int npad, int din, int Kpad) {
    long long total = (long long)npad * Kpad;
    long long stride = (long long)gridDim.x * blockDim.x;
    for (long long i = (long long)blockIdx.x * blockDim.x + threadIdx.x; i < total; i += stride) {
        int row = (int)(i / Kpad);
        int k   = (int)(i - (long long)row * Kpad);
        float v = 0.0f;
        if (row < n) {
            if (k < din)            v = msg[(size_t)row * din + k] * inv[row];
            else if (k < 2 * din)   v = x[(size_t)row * din + (k - din)];
        }
        Ah[i] = (_Float16)v;
    }
}

// Build f16 weight matrix: Wh[n][k] = k<din ? wl[n][k] : wr[n][k-din], zero padded.
__global__ __launch_bounds__(256) void build_w_kernel(const float* __restrict__ wl,
                                                      const float* __restrict__ wr,
                                                      _Float16* __restrict__ Wh,
                                                      int dout, int din, int Kpad) {
    int total = dout * Kpad;
    int stride = gridDim.x * blockDim.x;
    for (int i = blockIdx.x * blockDim.x + threadIdx.x; i < total; i += stride) {
        int nrow = i / Kpad;
        int k    = i - nrow * Kpad;
        float v = 0.0f;
        if (k < din)          v = wl[nrow * din + k];
        else if (k < 2 * din) v = wr[nrow * din + (k - din)];
        Wh[i] = (_Float16)v;
    }
}

// ------------------------------------------------- fused GEMM + bias + LN (+ReLU)
// Block = 128 threads = 4 waves; each wave computes 16 rows x DOUT cols via
// v_wmma_f32_16x16x32_f16 tiles. KPAD is a template parameter so the k-loop
// fully unrolls: all fragment loads become base+immediate-offset b128 loads
// that the scheduler can hoist across WMMAs (partial s_wait_loadcnt instead of
// a load->wait(0)->wmma serial chain per tile).
// LayerNorm is fused via half-wave shfl_xor reductions (each output row lives
// in 16 lanes under the WMMA C/D layout).
template <int NT, int KPAD, bool RELU>
__global__ __launch_bounds__(128) void gemm_ln_kernel(const _Float16* __restrict__ A,
                                                      const _Float16* __restrict__ W,
                                                      const float* __restrict__ bias,
                                                      const float* __restrict__ gamma,
                                                      const float* __restrict__ beta,
                                                      float* __restrict__ out,
                                                      int n) {
    constexpr int DOUT   = NT * 16;
    constexpr int KSTEPS = KPAD / 32;
    const int lane = threadIdx.x & 31;
    const int wid  = threadIdx.x >> 5;
    const int ln   = lane & 15;     // column / row-within-tile index
    const int hs   = lane >> 4;     // half-wave select
    const int rowBase = blockIdx.x * 64 + wid * 16;

    v8f acc[NT];
    #pragma unroll
    for (int t = 0; t < NT; ++t)
        #pragma unroll
        for (int r = 0; r < 8; ++r) acc[t][r] = 0.0f;

    // A fragment base: lane holds row rowBase+ln, 16 contiguous K per half-wave
    // (matching B).
    const _Float16* arow = A + (size_t)(rowBase + ln) * KPAD + hs * 16;
    // B fragment base: lane holds column t*16+ln, 16 contiguous K per half-wave.
    const _Float16* wrow = W + (size_t)ln * KPAD + hs * 16;

    #pragma unroll
    for (int ks = 0; ks < KSTEPS; ++ks) {
        const int k0 = ks * 32;
        if (ks + 1 < KSTEPS) __builtin_prefetch(arow + k0 + 32, 0, 3);

        v16h a = *(const v16h*)(arow + k0);         // K = k0+{0..15} (+16 upper half)

        #pragma unroll
        for (int t = 0; t < NT; ++t) {
            v16h b = *(const v16h*)(wrow + t * 16 * KPAD + k0);  // immediate offset
            acc[t] = __builtin_amdgcn_wmma_f32_16x16x32_f16(
                false, a, false, b, (short)0, acc[t], false, false);
        }
    }

    // epilogue: + bias, LayerNorm over DOUT, gamma/beta, optional ReLU
    float bl[NT], gm[NT], bt[NT];
    #pragma unroll
    for (int t = 0; t < NT; ++t) {
        int col = t * 16 + ln;
        bl[t] = bias[col];
        gm[t] = gamma[col];
        bt[t] = beta[col];
    }

    #pragma unroll
    for (int r = 0; r < 8; ++r) {
        float s = 0.0f;
        #pragma unroll
        for (int t = 0; t < NT; ++t) { acc[t][r] += bl[t]; s += acc[t][r]; }
        #pragma unroll
        for (int m = 1; m < 16; m <<= 1) s += __shfl_xor(s, m, 32);
        float mu = s * (1.0f / DOUT);

        float v = 0.0f;
        #pragma unroll
        for (int t = 0; t < NT; ++t) { float d = acc[t][r] - mu; v += d * d; }
        #pragma unroll
        for (int m = 1; m < 16; m <<= 1) v += __shfl_xor(v, m, 32);
        float rstd = rsqrtf(v * (1.0f / DOUT) + 1e-5f);

        int row = rowBase + r + hs * 8;   // D layout: M = r (+8 for upper half-wave)
        if (row < n) {
            #pragma unroll
            for (int t = 0; t < NT; ++t) {
                float val = (acc[t][r] - mu) * rstd * gm[t] + bt[t];
                if (RELU) val = fmaxf(val, 0.0f);
                out[(size_t)row * DOUT + t * 16 + ln] = val;
            }
        }
    }
}

// ---------------------------------------------------------------- host side

static void run_layer_pre(const float* xin, int din, int Kpad,
                          const float* wl, const float* wr, int dout,
                          float* inv_cnt, float* msg, _Float16* Ah, _Float16* Wh,
                          const int* src, const int* dst, hipStream_t stream) {
    const int TB = 256;
    zero_f32_kernel<<<2048, TB, 0, stream>>>(msg, (long long)NNODES * din);
    scatter_add_kernel<<<4096, TB, 0, stream>>>(xin, src, dst, msg,
                                                (long long)NEDGES * din, din);
    build_a_kernel<<<4096, TB, 0, stream>>>(msg, xin, inv_cnt, Ah,
                                            NNODES, NPAD, din, Kpad);
    build_w_kernel<<<64, TB, 0, stream>>>(wl, wr, Wh, dout, din, Kpad);
}

extern "C" void kernel_launch(void* const* d_in, const int* in_sizes, int n_in,
                              void* d_out, int out_size, void* d_ws, size_t ws_size,
                              hipStream_t stream) {
    (void)in_sizes; (void)n_in; (void)out_size; (void)ws_size;

    const float* x   = (const float*)d_in[0];
    const int*   ei  = (const int*)d_in[1];
    const int*   src = ei;
    const int*   dst = ei + NEDGES;

    // per-layer params: w{i}l, b{i}, w{i}r, g{i}, be{i} starting at index 2
    const float* wl[4]; const float* b[4]; const float* wr[4];
    const float* g[4];  const float* be[4];
    for (int i = 0; i < 4; ++i) {
        wl[i] = (const float*)d_in[2 + 5 * i + 0];
        b[i]  = (const float*)d_in[2 + 5 * i + 1];
        wr[i] = (const float*)d_in[2 + 5 * i + 2];
        g[i]  = (const float*)d_in[2 + 5 * i + 3];
        be[i] = (const float*)d_in[2 + 5 * i + 4];
    }

    // workspace carve (256B aligned)
    char* ws = (char*)d_ws;
    size_t off = 0;
    auto carve = [&](size_t bytes) -> char* {
        char* p = ws + off;
        off += (bytes + 255) & ~(size_t)255;
        return p;
    };
    float*    cnt = (float*)   carve((size_t)NNODES * sizeof(float));
    float*    msg = (float*)   carve((size_t)NNODES * 128 * sizeof(float));
    _Float16* Ah  = (_Float16*)carve((size_t)NPAD * 256 * sizeof(_Float16));
    _Float16* Wh  = (_Float16*)carve((size_t)128 * 256 * sizeof(_Float16));
    float*    h1  = (float*)   carve((size_t)NNODES * 128 * sizeof(float));
    float*    h2  = (float*)   carve((size_t)NNODES * 128 * sizeof(float));

    // degree counts (identical for all layers) -> inverse counts, in place
    zero_f32_kernel<<<1024, 256, 0, stream>>>(cnt, (long long)NNODES);
    count_kernel<<<4096, 256, 0, stream>>>(dst, cnt, NEDGES);
    inv_count_kernel<<<1024, 256, 0, stream>>>(cnt, NNODES);

    const int gb = NPAD / 64;

    // L1: 9 -> 64 (K=18 padded to 32), ReLU
    run_layer_pre(x, 9, 32, wl[0], wr[0], 64, cnt, msg, Ah, Wh, src, dst, stream);
    gemm_ln_kernel<4, 32, true><<<gb, 128, 0, stream>>>(
        Ah, Wh, b[0], g[0], be[0], h1, NNODES);

    // L2: 64 -> 128 (K=128), ReLU
    run_layer_pre(h1, 64, 128, wl[1], wr[1], 128, cnt, msg, Ah, Wh, src, dst, stream);
    gemm_ln_kernel<8, 128, true><<<gb, 128, 0, stream>>>(
        Ah, Wh, b[1], g[1], be[1], h2, NNODES);

    // L3: 128 -> 128 (K=256), ReLU
    run_layer_pre(h2, 128, 256, wl[2], wr[2], 128, cnt, msg, Ah, Wh, src, dst, stream);
    gemm_ln_kernel<8, 256, true><<<gb, 128, 0, stream>>>(
        Ah, Wh, b[2], g[2], be[2], h1, NNODES);

    // L4: 128 -> 128 (K=256), no ReLU, straight to output
    run_layer_pre(h1, 128, 256, wl[3], wr[3], 128, cnt, msg, Ah, Wh, src, dst, stream);
    gemm_ln_kernel<8, 256, false><<<gb, 128, 0, stream>>>(
        Ah, Wh, b[3], g[3], be[3], (float*)d_out, NNODES);
}
